// FeatureAlign_22376779612379
// MI455X (gfx1250) — compile-verified, matrix operand order checked by reading, not run on hardware
//
#include <hip/hip_runtime.h>
#include <hip/hip_bf16.h>
#include <math.h>

// ---------------- problem constants ----------------
#define HH   192
#define WW   192
#define HWPX (HH * WW)      // 36864
#define CC   64             // fea channels (== conv depth)
#define GN   4              // groups
#define PN   9              // kernel taps
#define GCN  16             // group channels
#define NCOMB 112           // 72 offset + 36 mask cols, padded to 16
#define EPSV 1e-5f

typedef __attribute__((ext_vector_type(16))) _Float16 v16h;
typedef __attribute__((ext_vector_type(8)))  float    v8f;
typedef __attribute__((ext_vector_type(8)))  _Float16 half8;
typedef int v4i_t __attribute__((vector_size(16)));   // matches builtin param type

// gfx1250 async global->LDS DMA (ASYNCcnt path), guarded so compile never breaks.
#if defined(__has_builtin)
#if __has_builtin(__builtin_amdgcn_global_load_async_to_lds_b128) && \
    __has_builtin(__builtin_amdgcn_s_wait_asynccnt)
#define USE_ASYNC_LDS 1
#endif
#endif
#ifndef USE_ASYNC_LDS
#define USE_ASYNC_LDS 0
#endif

// copy 16 bytes global -> LDS (per lane)
__device__ inline void copy16_g2lds(void* lds, const void* g) {
#if USE_ASYNC_LDS
    void* gp = (void*)g;  // strip const, then addrspace-cast to the builtin's types
    __builtin_amdgcn_global_load_async_to_lds_b128(
        (__attribute__((address_space(1))) v4i_t*)gp,
        (__attribute__((address_space(3))) v4i_t*)lds, 0, 0);
#else
    *(int4*)lds = *(const int4*)g;
#endif
}

__device__ inline void wait_async_copies() {
#if USE_ASYNC_LDS
    __builtin_amdgcn_s_wait_asynccnt(0);
#endif
}

// Load one 16x32 f16 WMMA operand fragment (A layout; B uses the same pattern
// against an n-major weight buffer). p = row_base + kb, already lane-adjusted.
// lane<16 : K = {0..7} u {16..23};  lane>=16 : K = {8..15} u {24..31}
__device__ inline v16h load_frag16(const _Float16* __restrict__ p) {
    half8 lo = *(const half8*)(p);        // K offsets +0 .. +7
    half8 hi = *(const half8*)(p + 16);   // K offsets +16 .. +23
    v16h r;
#pragma unroll
    for (int i = 0; i < 8; ++i) { r[i] = lo[i]; r[i + 8] = hi[i]; }
    return r;
}

// =====================================================================
// Kernel 1: fused alignConv  (conv3d 2->32 +BN+ReLU, conv3d 32->1 +BN+ReLU)
// Output tile per block: TD x TH x TW of the [D=64,H,W] map.
// conv1 halo tile kept in LDS; never materialized to HBM (would be 302 MB).
// Writes q in [H,W,C] layout (C == depth dim of the 3D conv).
// =====================================================================
#define TD 2
#define TH 4
#define TW 8

__global__ __launch_bounds__(256) void align_conv_kernel(
    const float* __restrict__ cur, const float* __restrict__ ref,
    const float* __restrict__ w1,
    const float* __restrict__ bn1g, const float* __restrict__ bn1b,
    const float* __restrict__ bn1m, const float* __restrict__ bn1v,
    const float* __restrict__ w2,
    const float* __restrict__ bn2g, const float* __restrict__ bn2b,
    const float* __restrict__ bn2m, const float* __restrict__ bn2v,
    float* __restrict__ qout)
{
    __shared__ float sIn[2 * 6 * 8 * 12];    // input tile  [ic][d 6][h 8][w 12]
    __shared__ float sC1[32 * 4 * 6 * 10];   // conv1 tile  [c1][d 4][h 6][w 10]
    __shared__ float sW1[32 * 54];
    __shared__ float sW2[32 * 27];
    __shared__ float sS1[32], sB1[32];

    const int tid = threadIdx.x;
    const int w0 = blockIdx.x * TW;
    const int h0 = blockIdx.y * TH;
    const int d0 = blockIdx.z * TD;

    for (int i = tid; i < 32 * 54; i += 256) sW1[i] = w1[i];
    for (int i = tid; i < 32 * 27; i += 256) sW2[i] = w2[i];
    if (tid < 32) {
        float s = bn1g[tid] * rsqrtf(bn1v[tid] + EPSV);
        sS1[tid] = s;
        sB1[tid] = bn1b[tid] - bn1m[tid] * s;
    }
    // input tile, origin (d0-2, h0-2, w0-2), zero padded
    for (int i = tid; i < 2 * 6 * 8 * 12; i += 256) {
        int ic = i / (6 * 8 * 12); int r = i % (6 * 8 * 12);
        int dd = r / (8 * 12); r %= (8 * 12);
        int hh = r / 12; int ww = r % 12;
        int d = d0 - 2 + dd, h = h0 - 2 + hh, w = w0 - 2 + ww;
        float v = 0.f;
        if (d >= 0 && d < CC && h >= 0 && h < HH && w >= 0 && w < WW) {
            const float* src = (ic == 0) ? cur : ref;
            v = src[(d * HH + h) * WW + w];
        }
        sIn[i] = v;
    }
    __syncthreads();

    // conv1 + BN1 + ReLU over halo tile, origin (d0-1, h0-1, w0-1)
    for (int o = tid; o < 32 * 240; o += 256) {
        int c1 = o / 240; int r = o % 240;
        int dd = r / 60; r %= 60;
        int hh = r / 10; int ww = r % 10;
        int d = d0 - 1 + dd, h = h0 - 1 + hh, w = w0 - 1 + ww;
        float acc = 0.f;
        if (d >= 0 && d < CC && h >= 0 && h < HH && w >= 0 && w < WW) {
#pragma unroll
            for (int ic = 0; ic < 2; ++ic)
                for (int kd = 0; kd < 3; ++kd)
                    for (int kh = 0; kh < 3; ++kh)
#pragma unroll
                        for (int kw = 0; kw < 3; ++kw)
                            acc += sIn[((ic * 6 + dd + kd) * 8 + hh + kh) * 12 + ww + kw]
                                 * sW1[c1 * 54 + ic * 27 + (kd * 3 + kh) * 3 + kw];
            acc = fmaxf(acc * sS1[c1] + sB1[c1], 0.f);
        }                                    // out-of-range -> conv2's zero pad
        sC1[((c1 * 4 + dd) * 6 + hh) * 10 + ww] = acc;
    }
    __syncthreads();

    // conv2 + BN2 + ReLU, one output per thread (64 outputs)
    if (tid < TD * TH * TW) {
        int dd = tid / (TH * TW); int r = tid % (TH * TW);
        int hh = r / TW; int ww = r % TW;
        float acc = 0.f;
        for (int c1 = 0; c1 < 32; ++c1)
#pragma unroll
            for (int kd = 0; kd < 3; ++kd)
                for (int kh = 0; kh < 3; ++kh)
#pragma unroll
                    for (int kw = 0; kw < 3; ++kw)
                        acc += sC1[((c1 * 4 + dd + kd) * 6 + hh + kh) * 10 + ww + kw]
                             * sW2[c1 * 27 + (kd * 3 + kh) * 3 + kw];
        float s2 = bn2g[0] * rsqrtf(bn2v[0] + EPSV);
        float b2 = bn2b[0] - bn2m[0] * s2;
        float y  = fmaxf(acc * s2 + b2, 0.f);
        int d = d0 + dd, h = h0 + hh, w = w0 + ww;
        qout[(h * WW + w) * CC + d] = y;
    }
}

// =====================================================================
// Kernel 2: depthwise 3x3 conv + LayerNorm(C=64) + exact GELU.
// One wave32 per pixel; lane owns channels {lane, lane+32}; wave32 butterfly
// reduction for mean/var. Emits f16 activations for the WMMA GEMMs.
// =====================================================================
__global__ __launch_bounds__(256) void dw_ln_gelu_kernel(
    const float* __restrict__ q,
    const float* __restrict__ dww, const float* __restrict__ dwb,
    const float* __restrict__ lng, const float* __restrict__ lnb,
    _Float16* __restrict__ x1h)
{
    int wave = threadIdx.x >> 5;
    int lane = threadIdx.x & 31;
    int pix  = blockIdx.x * 8 + wave;
    int h = pix / WW, w = pix % WW;

    float v[2];
#pragma unroll
    for (int half = 0; half < 2; ++half) {
        int c = lane + 32 * half;
        float acc = dwb[c];
#pragma unroll
        for (int ky = 0; ky < 3; ++ky)
#pragma unroll
            for (int kx = 0; kx < 3; ++kx) {
                int hy = h + ky - 1, wx = w + kx - 1;
                if (hy >= 0 && hy < HH && wx >= 0 && wx < WW)
                    acc += q[(hy * WW + wx) * CC + c] * dww[c * 9 + ky * 3 + kx];
            }
        v[half] = acc;
    }
    float s1 = v[0] + v[1];
    float s2 = v[0] * v[0] + v[1] * v[1];
#pragma unroll
    for (int off = 16; off > 0; off >>= 1) {   // wave32 butterfly
        s1 += __shfl_xor(s1, off, 32);
        s2 += __shfl_xor(s2, off, 32);
    }
    float mu  = s1 * (1.f / 64.f);
    float var = s2 * (1.f / 64.f) - mu * mu;
    float inv = rsqrtf(var + EPSV);
#pragma unroll
    for (int half = 0; half < 2; ++half) {
        int c = lane + 32 * half;
        float x = (v[half] - mu) * inv * lng[c] + lnb[c];
        x = 0.5f * x * (1.f + erff(x * 0.70710678f));   // exact GELU
        x1h[pix * CC + c] = (_Float16)x;
    }
}

// =====================================================================
// Kernel 3: weight prep — f16 n-major copies of [off_w|mask_w|pad] and out_w,
// combined bias vector.
// =====================================================================
__global__ __launch_bounds__(256) void prep_weights_kernel(
    const float* __restrict__ offw, const float* __restrict__ offb,
    const float* __restrict__ maskw, const float* __restrict__ maskb,
    const float* __restrict__ outw,
    _Float16* __restrict__ Wcomb, float* __restrict__ bcomb,
    _Float16* __restrict__ WoutT)
{
    int tid = blockIdx.x * 256 + threadIdx.x;
    if (tid < NCOMB * 64) {
        int n = tid / 64, k = tid % 64;
        float v = 0.f;
        if (n < 72)       v = offw[k * 72 + n];
        else if (n < 108) v = maskw[k * 36 + (n - 72)];
        Wcomb[n * 64 + k] = (_Float16)v;
    }
    if (tid < NCOMB) {
        float b = 0.f;
        if (tid < 72)       b = offb[tid];
        else if (tid < 108) b = maskb[tid - 72];
        bcomb[tid] = b;
    }
    if (tid < 64 * 64) {
        int n = tid / 64, k = tid % 64;
        WoutT[n * 64 + k] = (_Float16)outw[k * 64 + n];
    }
}

// =====================================================================
// Kernel 4: reference feature CHW -> HWC (grouped layout for the sampler)
// =====================================================================
__global__ __launch_bounds__(256) void transpose_ref_kernel(
    const float* __restrict__ ref, float* __restrict__ refHWC)
{
    int tid = blockIdx.x * 256 + threadIdx.x;   // HW * 64 threads
    int pix = tid / CC, c = tid % CC;
    refHWC[tid] = ref[c * HWPX + pix];
}

// =====================================================================
// Kernel 5: K=64 GEMM on v_wmma_f32_16x16x32_f16.
// A: [M,64] f16 row-major.  Bt: [N,64] f16 (n-major, i.e. B pre-transposed).
// The whole weight matrix (<= 14 KB) is staged into LDS once per block with
// gfx1250 async global->LDS DMA; each wave owns one 16-row m-tile, loads its
// A fragments once and reuses them across all n-tiles (B served from LDS).
// Epilogue adds bias, stores C[m*ldc+n] or transposed C[n*M+m].
// =====================================================================
__global__ __launch_bounds__(256) void gemm_k64_wmma_kernel(
    const _Float16* __restrict__ A, const _Float16* __restrict__ Bt,
    const float* __restrict__ bias, float* __restrict__ C,
    int M, int ntiles, int ldc, int transposed_store)
{
    __shared__ _Float16 sBt[NCOMB * 64];         // n-major weights, <=14 KB
    const int wave = threadIdx.x >> 5;
    const int lane = threadIdx.x & 31;

    // ---- stage weights into LDS (ASYNCcnt DMA path; 512 B per wave-issue)
    const int totalB = ntiles << 11;             // ntiles * 16 rows * 64 * 2B
    for (int ofs = wave * 512 + lane * 16; ofs < totalB; ofs += 8 * 512)
        copy16_g2lds((char*)sBt + ofs, (const char*)Bt + ofs);
    wait_async_copies();
    __syncthreads();

    // ---- A fragments: loaded once, reused for every n-tile
    int mt = blockIdx.x * 8 + wave;              // m-tile of this wave
    if (mt >= (M >> 4)) return;                  // wave-uniform guard
    int m0 = mt << 4;
    int kb = (lane >> 4) << 3;                   // +8 for upper half-wave
    const _Float16* pa = A + (m0 + (lane & 15)) * 64 + kb;
    __builtin_prefetch(pa + 128 * 64, 0, 1);     // next block's A -> global_prefetch_b8
    v16h a0 = load_frag16(pa);                   // K = 0..31
    v16h a1 = load_frag16(pa + 32);              // K = 32..63
    int mrow = m0 + ((lane >> 4) << 3);          // VGPR r -> M = mrow + r

    for (int tn = 0; tn < ntiles; ++tn) {
        const _Float16* pb = sBt + (tn * 16 + (lane & 15)) * 64 + kb;
        v16h b0 = load_frag16(pb);               // ds_load_b128 x2
        v16h b1 = load_frag16(pb + 32);
        v8f acc = {};
        acc = __builtin_amdgcn_wmma_f32_16x16x32_f16(
                  false, a0, false, b0, (short)0, acc, false, false);
        acc = __builtin_amdgcn_wmma_f32_16x16x32_f16(
                  false, a1, false, b1, (short)0, acc, false, false);
        int n = tn * 16 + (lane & 15);
        float bn = bias[n];
        if (transposed_store) {
#pragma unroll
            for (int r = 0; r < 8; ++r) C[n * M + (mrow + r)] = acc[r] + bn;
        } else {
#pragma unroll
            for (int r = 0; r < 8; ++r) C[(mrow + r) * ldc + n] = acc[r] + bn;
        }
    }
}

// =====================================================================
// Kernel 6: DCNv3 core — per (pixel, group): softmax over 9 taps, bilinear
// gather of 16 group-channels (4x float4) from the HWC reference map
// (9.4 MB -> resident in the 192 MB L2).
// Padded-map coords: px = w + ix + off_x  (w+1.5 + (ix-1) + off - 0.5).
// =====================================================================
__global__ __launch_bounds__(256) void dcn_sample_kernel(
    const float* __restrict__ refHWC, const float* __restrict__ gout,
    _Float16* __restrict__ dcnh)
{
    int gid = blockIdx.x * 256 + threadIdx.x;    // HW * G threads
    int g   = gid & 3;
    int pix = gid >> 2;
    int h = pix / WW, w = pix % WW;
    const float* row = gout + pix * NCOMB;

    float lg[PN];
    float mx = -1e30f;
#pragma unroll
    for (int p = 0; p < PN; ++p) { lg[p] = row[72 + g * PN + p]; mx = fmaxf(mx, lg[p]); }
    float ssum = 0.f;
#pragma unroll
    for (int p = 0; p < PN; ++p) { lg[p] = __expf(lg[p] - mx); ssum += lg[p]; }
    float rs = 1.f / ssum;

    float4 acc0 = {0.f, 0.f, 0.f, 0.f}, acc1 = acc0, acc2 = acc0, acc3 = acc0;

#pragma unroll
    for (int p = 0; p < PN; ++p) {
        float m  = lg[p] * rs;
        float ox = row[(g * PN + p) * 2 + 0];
        float oy = row[(g * PN + p) * 2 + 1];
        float px = (float)(w + (p / 3)) + ox;    // padded-map x
        float py = (float)(h + (p % 3)) + oy;    // padded-map y
        float x0f = floorf(px), y0f = floorf(py);
        float wx = px - x0f, wy = py - y0f;
        int x0 = (int)x0f, y0 = (int)y0f;
#pragma unroll
        for (int cy = 0; cy < 2; ++cy)
#pragma unroll
            for (int cx = 0; cx < 2; ++cx) {
                int xi = x0 + cx - 1;            // image coords (pad ring = 0)
                int yi = y0 + cy - 1;
                float wgt = m * (cx ? wx : 1.f - wx) * (cy ? wy : 1.f - wy);
                if (xi >= 0 && xi < WW && yi >= 0 && yi < HH) {
                    const float4* src =
                        (const float4*)(refHWC + (yi * WW + xi) * CC + g * GCN);
                    float4 t0 = src[0], t1 = src[1], t2 = src[2], t3 = src[3];
                    acc0.x += wgt * t0.x; acc0.y += wgt * t0.y;
                    acc0.z += wgt * t0.z; acc0.w += wgt * t0.w;
                    acc1.x += wgt * t1.x; acc1.y += wgt * t1.y;
                    acc1.z += wgt * t1.z; acc1.w += wgt * t1.w;
                    acc2.x += wgt * t2.x; acc2.y += wgt * t2.y;
                    acc2.z += wgt * t2.z; acc2.w += wgt * t2.w;
                    acc3.x += wgt * t3.x; acc3.y += wgt * t3.y;
                    acc3.z += wgt * t3.z; acc3.w += wgt * t3.w;
                }
            }
    }
    _Float16* dst = dcnh + pix * CC + g * GCN;
    float af[GCN] = {acc0.x, acc0.y, acc0.z, acc0.w, acc1.x, acc1.y, acc1.z, acc1.w,
                     acc2.x, acc2.y, acc2.z, acc2.w, acc3.x, acc3.y, acc3.z, acc3.w};
#pragma unroll
    for (int i = 0; i < GCN; ++i) dst[i] = (_Float16)af[i];
}

// =====================================================================
// Host-side orchestration
// =====================================================================
extern "C" void kernel_launch(void* const* d_in, const int* in_sizes, int n_in,
                              void* d_out, int out_size, void* d_ws, size_t ws_size,
                              hipStream_t stream)
{
    (void)in_sizes; (void)n_in; (void)out_size; (void)ws_size;
    const float* cur   = (const float*)d_in[0];
    const float* ref   = (const float*)d_in[1];
    const float* w1    = (const float*)d_in[2];
    const float* bn1g  = (const float*)d_in[3];
    const float* bn1b  = (const float*)d_in[4];
    const float* bn1m  = (const float*)d_in[5];
    const float* bn1v  = (const float*)d_in[6];
    const float* w2    = (const float*)d_in[7];
    const float* bn2g  = (const float*)d_in[8];
    const float* bn2b  = (const float*)d_in[9];
    const float* bn2m  = (const float*)d_in[10];
    const float* bn2v  = (const float*)d_in[11];
    const float* dww   = (const float*)d_in[12];
    const float* dwb   = (const float*)d_in[13];
    const float* lng   = (const float*)d_in[14];
    const float* lnb   = (const float*)d_in[15];
    const float* offw  = (const float*)d_in[16];
    const float* offb  = (const float*)d_in[17];
    const float* maskw = (const float*)d_in[18];
    const float* maskb = (const float*)d_in[19];
    const float* outw  = (const float*)d_in[20];
    const float* outb  = (const float*)d_in[21];

    char* ws = (char*)d_ws;
    size_t off = 0;
    float*    qbuf   = (float*)(ws + off);    off += (size_t)HWPX * CC * 4;      // 9.4 MB
    _Float16* x1h    = (_Float16*)(ws + off); off += (size_t)HWPX * CC * 2;      // 4.7 MB
    float*    g1out  = (float*)(ws + off);    off += (size_t)HWPX * NCOMB * 4;   // 16.5 MB
    float*    refHWC = (float*)(ws + off);    off += (size_t)HWPX * CC * 4;      // 9.4 MB
    _Float16* dcnh   = (_Float16*)(ws + off); off += (size_t)HWPX * CC * 2;      // 4.7 MB
    _Float16* Wcomb  = (_Float16*)(ws + off); off += (size_t)NCOMB * 64 * 2;
    _Float16* WoutT  = (_Float16*)(ws + off); off += (size_t)64 * 64 * 2;
    float*    bcomb  = (float*)(ws + off);    off += 512;

    // 1) weight prep + reference transpose (independent)
    prep_weights_kernel<<<(NCOMB * 64 + 255) / 256, 256, 0, stream>>>(
        offw, offb, maskw, maskb, outw, Wcomb, bcomb, WoutT);
    transpose_ref_kernel<<<(HWPX * CC) / 256, 256, 0, stream>>>(ref, refHWC);

    // 2) fused alignConv stack -> q [H,W,C]
    dim3 gAlign(WW / TW, HH / TH, CC / TD);   // 24 x 48 x 32
    align_conv_kernel<<<gAlign, 256, 0, stream>>>(
        cur, ref, w1, bn1g, bn1b, bn1m, bn1v, w2, bn2g, bn2b, bn2m, bn2v, qbuf);

    // 3) depthwise + LN + GELU -> x1 (f16)
    dw_ln_gelu_kernel<<<HWPX / 8, 256, 0, stream>>>(qbuf, dww, dwb, lng, lnb, x1h);

    // 4) offset+mask GEMM (WMMA): [36864,64] x [64,112] -> g1out
    gemm_k64_wmma_kernel<<<(HWPX / 16) / 8, 256, 0, stream>>>(
        x1h, Wcomb, bcomb, g1out, HWPX, NCOMB / 16, NCOMB, 0);

    // 5) DCNv3 sampling -> dcnh (f16)
    dcn_sample_kernel<<<(HWPX * GN) / 256, 256, 0, stream>>>(refHWC, g1out, dcnh);

    // 6) output projection GEMM (WMMA), transposed store into [C,H,W] d_out
    gemm_k64_wmma_kernel<<<(HWPX / 16) / 8, 256, 0, stream>>>(
        dcnh, WoutT, outb, (float*)d_out, HWPX, CC / 16, CC, 1);
}
